// TransformerLayer_7370163880222
// MI455X (gfx1250) — matrix operand order repaired and checked by reading, not compiled, not run on hardware
//
#include <hip/hip_runtime.h>
#include <hip/hip_bf16.h>

// ---------------------------------------------------------------------------
// Fused transformer encoder layer for MI455X (gfx1250, wave32, WMMA).
// One workgroup (256 thr = 8 waves) per sequence (b*N = 1024 sequences).
// All intermediates in LDS; matmuls via v_wmma_f32_16x16x32_f16.
// ---------------------------------------------------------------------------

typedef __attribute__((ext_vector_type(16))) _Float16 v16h;
typedef __attribute__((ext_vector_type(8)))  float    v8f;
typedef __attribute__((ext_vector_type(4)))  float    f32x4;

union Frag16 { v16h v; _Float16 h[16]; f32x4 q[2]; };
union AccF   { v8f  v; float f[8]; };

#define D_MODEL 128
#define D_FF    512
#define XLD     136            // padded row pitch (halves): 272B rows, 16B aligned
#define Y1LD    520            // padded pitch for 512-wide FFN intermediate
#define HB      (128 * XLD)    // halves per 128x136 buffer = 17408

// LDS half-offsets
#define OFF_X   0
#define OFF_Q   (1 * HB)
#define OFF_K   (2 * HB)
#define OFF_VT  (3 * HB)
#define OFF_P   (4 * HB)
#define OFF_O   (5 * HB)
#define OFF_XN  (6 * HB)
#define CONST_BYTE_OFF (7 * HB * 2)
#define N_CONST 1664
#define SMEM_BYTES (CONST_BYTE_OFF + N_CONST * 4)

// const-table float offsets
#define CB_BQ   0
#define CB_BK   128
#define CB_BV   256
#define CB_BO   384
#define CB_B1   512
#define CB_B2   1024
#define CB_G1   1152
#define CB_BE1  1280
#define CB_G2   1408
#define CB_BE2  1536

// f16 weight segments inside d_ws (halves)
#define WS_WQ   0
#define WS_WK   16384
#define WS_WV   32768
#define WS_WO   49152
#define WS_W1   65536
#define WS_W2   131072
#define WS_TOTAL 196608

__device__ __forceinline__ f32x4 f4zero() {
    f32x4 z; z.x = 0.f; z.y = 0.f; z.z = 0.f; z.w = 0.f; return z;
}

__device__ __forceinline__ v8f wmma_f16(v16h a, v16h b, v8f c) {
    // v_wmma_f32_16x16x32_f16  D = A*B + C
    return __builtin_amdgcn_wmma_f32_16x16x32_f16(false, a, false, b,
                                                  (short)0, c, false, false);
}

// A fragment (16x32 f16) from a row-major matrix, row pitch `ld` halves.
// Lane L: row = row0 + (L&15); half h=L>>4: regs0-3 = K k0+8h..+7, regs4-7 = +16.
__device__ __forceinline__ Frag16 loadA(const _Float16* m, int ld, int row0, int k0) {
    const int lane = threadIdx.x & 31;
    const int mr = lane & 15, hh = lane >> 4;
    const _Float16* p = m + (row0 + mr) * ld + k0 + hh * 8;
    Frag16 a;
    a.q[0] = *(const f32x4*)p;
    a.q[1] = *(const f32x4*)(p + 16);
    return a;
}

// B fragment (32x16) where B[k][n] = M[n0+n][k], M row-major pitch `ld` halves.
// Lane L: col n = L&15; half h=L>>4: regs0-3 = K k0+8h..+7, regs4-7 = +16.
__device__ __forceinline__ Frag16 loadB(const _Float16* m, int ld, int n0, int k0) {
    const int lane = threadIdx.x & 31;
    const int n = lane & 15, hh = lane >> 4;
    const _Float16* p = m + (n0 + n) * ld + k0 + hh * 8;
    Frag16 b;
    b.q[0] = *(const f32x4*)p;
    b.q[1] = *(const f32x4*)(p + 16);
    return b;
}

// ---------------------------------------------------------------------------
// Weight fp32 -> f16 conversion (runs once per launch; weights are tiny).
// ---------------------------------------------------------------------------
__global__ void convert_weights_kernel(const float* __restrict__ Wq,
                                       const float* __restrict__ Wk,
                                       const float* __restrict__ Wv,
                                       const float* __restrict__ Wo,
                                       const float* __restrict__ W1,
                                       const float* __restrict__ W2,
                                       _Float16* __restrict__ wh) {
    int i = blockIdx.x * 256 + threadIdx.x;
    if (i >= WS_TOTAL) return;
    float v;
    if      (i < WS_WK) v = Wq[i - WS_WQ];
    else if (i < WS_WV) v = Wk[i - WS_WK];
    else if (i < WS_WO) v = Wv[i - WS_WV];
    else if (i < WS_W1) v = Wo[i - WS_WO];
    else if (i < WS_W2) v = W1[i - WS_W1];
    else                v = W2[i - WS_W2];
    wh[i] = (_Float16)v;
}

// ---------------------------------------------------------------------------
// Fused layer kernel.
// ---------------------------------------------------------------------------
__global__ void __launch_bounds__(256)
transformer_layer_kernel(const float* __restrict__ x,
                         const _Float16* __restrict__ wh,
                         const float* __restrict__ bq, const float* __restrict__ bk,
                         const float* __restrict__ bv, const float* __restrict__ bo,
                         const float* __restrict__ b1, const float* __restrict__ b2,
                         const float* __restrict__ g1, const float* __restrict__ be1,
                         const float* __restrict__ g2, const float* __restrict__ be2,
                         float* __restrict__ out) {
    extern __shared__ char smem[];
    _Float16* Xs  = (_Float16*)smem;
    _Float16* Qs  = Xs + OFF_Q;
    _Float16* Ks  = Xs + OFF_K;
    _Float16* VTs = Xs + OFF_VT;
    _Float16* Ps  = Xs + OFF_P;
    _Float16* Os  = Xs + OFF_O;
    _Float16* XNs = Xs + OFF_XN;
    float*    cbt = (float*)(smem + CONST_BYTE_OFF);

    const int tid  = threadIdx.x;
    const int lane = tid & 31;
    const int wid  = tid >> 5;
    const int row0 = wid * 16;
    const int hh   = lane >> 4;
    const int ln   = lane & 15;

    const int seq  = blockIdx.x;          // 0..1023 = b*128 + n
    const int bIdx = seq >> 7;
    const int nIdx = seq & 127;
    const float* xg = x + (size_t)bIdx * 2097152 + (size_t)nIdx * 128;

    // --- stage consts + input tile into LDS (coalesced along t) ------------
    for (int i = tid; i < 128; i += 256) {
        cbt[CB_BQ + i] = bq[i];  cbt[CB_BK + i]  = bk[i];
        cbt[CB_BV + i] = bv[i];  cbt[CB_BO + i]  = bo[i];
        cbt[CB_B2 + i] = b2[i];  cbt[CB_G1 + i]  = g1[i];
        cbt[CB_BE1 + i] = be1[i]; cbt[CB_G2 + i] = g2[i];
        cbt[CB_BE2 + i] = be2[i];
    }
    for (int i = tid; i < 512; i += 256) cbt[CB_B1 + i] = b1[i];

    for (int idx = tid; idx < 128 * 128; idx += 256) {
        int c = idx >> 7, t = idx & 127;
        Xs[t * XLD + c] = (_Float16)xg[(size_t)c * 16384 + t];
    }
    __syncthreads();

    // --- step 1: Q,K,V projections (shared A fragments) --------------------
    {
        Frag16 aF[4];
#pragma unroll
        for (int kk = 0; kk < 4; kk++) aF[kk] = loadA(Xs, XLD, row0, kk * 32);
#pragma unroll 1
        for (int jt = 0; jt < 8; jt++) {
            AccF aq, ak, av;
#pragma unroll
            for (int i = 0; i < 8; i++) { aq.f[i] = 0.f; ak.f[i] = 0.f; av.f[i] = 0.f; }
#pragma unroll
            for (int kk = 0; kk < 4; kk++) {
                Frag16 bWq = loadB(wh + WS_WQ, 128, jt * 16, kk * 32);
                Frag16 bWk = loadB(wh + WS_WK, 128, jt * 16, kk * 32);
                Frag16 bWv = loadB(wh + WS_WV, 128, jt * 16, kk * 32);
                aq.v = wmma_f16(aF[kk].v, bWq.v, aq.v);
                ak.v = wmma_f16(aF[kk].v, bWk.v, ak.v);
                av.v = wmma_f16(aF[kk].v, bWv.v, av.v);
            }
            int c = jt * 16 + ln;
            float cbq = cbt[CB_BQ + c], cbk = cbt[CB_BK + c], cbv = cbt[CB_BV + c];
#pragma unroll
            for (int r = 0; r < 8; r++) {
                int row = row0 + hh * 8 + r;
                Qs[row * XLD + c]  = (_Float16)(aq.f[r] + cbq);
                Ks[row * XLD + c]  = (_Float16)(ak.f[r] + cbk);
                VTs[c * XLD + row] = (_Float16)(av.f[r] + cbv);   // transposed V
            }
        }
    }
    __syncthreads();

    // --- step 2: attention, head by head (E=16 padded to K=32) -------------
    const float scl = 0.25f;  // 1/sqrt(E)
#pragma unroll 1
    for (int h = 0; h < 8; h++) {
        Frag16 aq;
        {
            const _Float16* p = Qs + (row0 + ln) * XLD + h * 16 + hh * 8;
            aq.q[0] = *(const f32x4*)p;   // K=0..15 real
            aq.q[1] = f4zero();           // K=16..31 zero pad
        }
        AccF s[8];
#pragma unroll
        for (int jt = 0; jt < 8; jt++) {
            Frag16 bkf;
            {
                const _Float16* p = Ks + (jt * 16 + ln) * XLD + h * 16 + hh * 8;
                bkf.q[0] = *(const f32x4*)p;
                bkf.q[1] = f4zero();
            }
            AccF z;
#pragma unroll
            for (int i = 0; i < 8; i++) z.f[i] = 0.f;
            s[jt].v = wmma_f16(aq.v, bkf.v, z.v);
        }
        // row-wise softmax in C-register layout
#pragma unroll
        for (int r = 0; r < 8; r++) {
            float mx = -3.402823466e38f;
#pragma unroll
            for (int jt = 0; jt < 8; jt++) mx = fmaxf(mx, s[jt].f[r]);
#pragma unroll
            for (int m = 1; m < 16; m <<= 1) mx = fmaxf(mx, __shfl_xor(mx, m, 32));
            float sum = 0.f;
#pragma unroll
            for (int jt = 0; jt < 8; jt++) {
                float e = __expf((s[jt].f[r] - mx) * scl);
                s[jt].f[r] = e; sum += e;
            }
#pragma unroll
            for (int m = 1; m < 16; m <<= 1) sum += __shfl_xor(sum, m, 32);
            float inv = 1.0f / sum;
            int row = row0 + hh * 8 + r;
#pragma unroll
            for (int jt = 0; jt < 8; jt++)
                Ps[row * XLD + jt * 16 + ln] = (_Float16)(s[jt].f[r] * inv);
        }
        // O_h = P @ V_h  (B frags contiguous thanks to transposed V)
        AccF o;
#pragma unroll
        for (int i = 0; i < 8; i++) o.f[i] = 0.f;
#pragma unroll
        for (int kk = 0; kk < 4; kk++) {
            Frag16 ap  = loadA(Ps, XLD, row0, kk * 32);
            Frag16 bvf = loadB(VTs + h * 16 * XLD, XLD, 0, kk * 32);
            o.v = wmma_f16(ap.v, bvf.v, o.v);
        }
#pragma unroll
        for (int r = 0; r < 8; r++)
            Os[(row0 + hh * 8 + r) * XLD + h * 16 + ln] = (_Float16)o.f[r];
    }

    // --- step 3: Wo projection + residual + LayerNorm1 ---------------------
    AccF xr[8];
    {
        Frag16 aO[4];
#pragma unroll
        for (int kk = 0; kk < 4; kk++) aO[kk] = loadA(Os, XLD, row0, kk * 32);
#pragma unroll 1
        for (int jt = 0; jt < 8; jt++) {
            AccF acc;
#pragma unroll
            for (int i = 0; i < 8; i++) acc.f[i] = 0.f;
#pragma unroll
            for (int kk = 0; kk < 4; kk++) {
                Frag16 b = loadB(wh + WS_WO, 128, jt * 16, kk * 32);
                acc.v = wmma_f16(aO[kk].v, b.v, acc.v);
            }
            int c = jt * 16 + ln;
            float bob = cbt[CB_BO + c];
#pragma unroll
            for (int r = 0; r < 8; r++) {
                int row = row0 + hh * 8 + r;
                xr[jt].f[r] = acc.f[r] + bob + (float)Xs[row * XLD + c];
            }
        }
    }
#pragma unroll
    for (int r = 0; r < 8; r++) {
        float sum = 0.f, sq = 0.f;
#pragma unroll
        for (int jt = 0; jt < 8; jt++) { float v = xr[jt].f[r]; sum += v; sq += v * v; }
#pragma unroll
        for (int m = 1; m < 16; m <<= 1) {
            sum += __shfl_xor(sum, m, 32);
            sq  += __shfl_xor(sq,  m, 32);
        }
        float mean = sum * (1.0f / 128.0f);
        float var  = sq * (1.0f / 128.0f) - mean * mean;
        float rs   = rsqrtf(var + 1e-5f);
        int row = row0 + hh * 8 + r;
#pragma unroll
        for (int jt = 0; jt < 8; jt++) {
            int c = jt * 16 + ln;
            float v = (xr[jt].f[r] - mean) * rs * cbt[CB_G1 + c] + cbt[CB_BE1 + c];
            XNs[row * XLD + c] = (_Float16)v;
        }
    }
    __syncthreads();   // Q/K/VT/P region about to be reused as FFN intermediate

    // --- step 4: FFN up (128 -> 512) + ReLU --------------------------------
    _Float16* Y1 = Qs;   // reuse 4 buffers as 128 x 520 f16 region
    {
        Frag16 aN[4];
#pragma unroll
        for (int kk = 0; kk < 4; kk++) aN[kk] = loadA(XNs, XLD, row0, kk * 32);
#pragma unroll 1
        for (int jt = 0; jt < 32; jt++) {
            AccF acc;
#pragma unroll
            for (int i = 0; i < 8; i++) acc.f[i] = 0.f;
#pragma unroll
            for (int kk = 0; kk < 4; kk++) {
                Frag16 b = loadB(wh + WS_W1, 128, jt * 16, kk * 32);
                acc.v = wmma_f16(aN[kk].v, b.v, acc.v);
            }
            int f = jt * 16 + ln;
            float b1v = cbt[CB_B1 + f];
#pragma unroll
            for (int r = 0; r < 8; r++) {
                float v = fmaxf(acc.f[r] + b1v, 0.0f);
                Y1[(row0 + hh * 8 + r) * Y1LD + f] = (_Float16)v;
            }
        }
    }

    // --- step 5: FFN down (512 -> 128) + residual + LayerNorm2 -------------
    AccF acc2[8];
#pragma unroll
    for (int jt = 0; jt < 8; jt++)
#pragma unroll
        for (int i = 0; i < 8; i++) acc2[jt].f[i] = 0.f;
#pragma unroll 1
    for (int kk = 0; kk < 16; kk++) {
        Frag16 a = loadA(Y1, Y1LD, row0, kk * 32);
#pragma unroll
        for (int jt = 0; jt < 8; jt++) {
            Frag16 b = loadB(wh + WS_W2, 512, jt * 16, kk * 32);
            acc2[jt].v = wmma_f16(a.v, b.v, acc2[jt].v);
        }
    }
#pragma unroll
    for (int jt = 0; jt < 8; jt++) {
        int c = jt * 16 + ln;
        float b2v = cbt[CB_B2 + c];
#pragma unroll
        for (int r = 0; r < 8; r++) {
            int row = row0 + hh * 8 + r;
            acc2[jt].f[r] += b2v + (float)XNs[row * XLD + c];
        }
    }
#pragma unroll
    for (int r = 0; r < 8; r++) {
        float sum = 0.f, sq = 0.f;
#pragma unroll
        for (int jt = 0; jt < 8; jt++) { float v = acc2[jt].f[r]; sum += v; sq += v * v; }
#pragma unroll
        for (int m = 1; m < 16; m <<= 1) {
            sum += __shfl_xor(sum, m, 32);
            sq  += __shfl_xor(sq,  m, 32);
        }
        float mean = sum * (1.0f / 128.0f);
        float var  = sq * (1.0f / 128.0f) - mean * mean;
        float rs   = rsqrtf(var + 1e-5f);
#pragma unroll
        for (int jt = 0; jt < 8; jt++) {
            int c = jt * 16 + ln;
            acc2[jt].f[r] = (acc2[jt].f[r] - mean) * rs * cbt[CB_G2 + c] + cbt[CB_BE2 + c];
        }
    }

    // --- step 6: transpose through LDS, coalesced store to (b,C,N,T) -------
    __syncthreads();   // everyone done reading Y1 region before reuse as f32
    float* OT = (float*)(smem + OFF_Q * 2);   // 128x128 f32 staging
#pragma unroll
    for (int jt = 0; jt < 8; jt++) {
        int c = jt * 16 + ln;
#pragma unroll
        for (int r = 0; r < 8; r++) {
            int row = row0 + hh * 8 + r;
            OT[c * 128 + row] = acc2[jt].f[r];
        }
    }
    __syncthreads();
    float* og = out + (size_t)bIdx * 2097152 + (size_t)nIdx * 128;
    for (int idx = tid; idx < 128 * 128; idx += 256) {
        int c = idx >> 7, t = idx & 127;
        og[(size_t)c * 16384 + t] = OT[idx];
    }
}

// ---------------------------------------------------------------------------
extern "C" void kernel_launch(void* const* d_in, const int* in_sizes, int n_in,
                              void* d_out, int out_size, void* d_ws, size_t ws_size,
                              hipStream_t stream) {
    (void)in_sizes; (void)n_in; (void)out_size; (void)ws_size;
    const float* x   = (const float*)d_in[0];
    const float* Wq  = (const float*)d_in[1];
    const float* bq  = (const float*)d_in[2];
    const float* Wk  = (const float*)d_in[3];
    const float* bk  = (const float*)d_in[4];
    const float* Wv  = (const float*)d_in[5];
    const float* bv  = (const float*)d_in[6];
    const float* Wo  = (const float*)d_in[7];
    const float* bo  = (const float*)d_in[8];
    const float* W1  = (const float*)d_in[9];
    const float* b1  = (const float*)d_in[10];
    const float* W2  = (const float*)d_in[11];
    const float* b2  = (const float*)d_in[12];
    const float* g1  = (const float*)d_in[13];
    const float* be1 = (const float*)d_in[14];
    const float* g2  = (const float*)d_in[15];
    const float* be2 = (const float*)d_in[16];
    _Float16* wh = (_Float16*)d_ws;

    convert_weights_kernel<<<WS_TOTAL / 256, 256, 0, stream>>>(Wq, Wk, Wv, Wo, W1, W2, wh);

    hipFuncSetAttribute(reinterpret_cast<const void*>(transformer_layer_kernel),
                        hipFuncAttributeMaxDynamicSharedMemorySize, SMEM_BYTES);
    transformer_layer_kernel<<<1024, 256, SMEM_BYTES, stream>>>(
        x, wh, bq, bk, bv, bo, b1, b2, g1, be1, g2, be2, (float*)d_out);
}
